// mindLayer_28243704939342
// MI455X (gfx1250) — matrix operand order repaired and checked by reading, not compiled, not run on hardware
//
#include <hip/hip_runtime.h>
#include <hip/hip_bf16.h>
#include <math.h>

typedef __attribute__((ext_vector_type(16))) _Float16 v16h;
typedef __attribute__((ext_vector_type(8)))  _Float16 v8h;
typedef __attribute__((ext_vector_type(8)))  float    v8f;

#define FEAT 128

// ---------------------------------------------------------------------------
// prep: xa (f16), W transposed (f16), acc = b_local + b_global, deg = 1.0
// ---------------------------------------------------------------------------
__global__ __launch_bounds__(256) void prep_kernel(
    const float* __restrict__ act, const float* __restrict__ lrn,
    const float* __restrict__ Wl,  const float* __restrict__ Wg,
    const float* __restrict__ bl,  const float* __restrict__ bg,
    _Float16* __restrict__ xa_h,
    _Float16* __restrict__ Wl_t, _Float16* __restrict__ Wg_t,
    float* __restrict__ acc, float* __restrict__ deg_l, float* __restrict__ deg_g,
    int n) {
    const size_t i = (size_t)blockIdx.x * blockDim.x + threadIdx.x;
    const size_t total = (size_t)n * FEAT;
    if (i < total) {
        xa_h[i] = (_Float16)(act[i] + lrn[i]);
        const int c = (int)(i & (FEAT - 1));
        acc[i] = bl[c] + bg[c];
    }
    if (i < (size_t)n) {
        deg_l[i] = 1.0f;   // self-loop weight
        deg_g[i] = 1.0f;
    }
    if (i < (size_t)(FEAT * FEAT)) {
        const int k = (int)(i >> 7);
        const int c = (int)(i & (FEAT - 1));
        Wl_t[(size_t)c * FEAT + k] = (_Float16)Wl[i];
        Wg_t[(size_t)c * FEAT + k] = (_Float16)Wg[i];
    }
}

// ---------------------------------------------------------------------------
// degree scatter: deg[dst] += w_e
// ---------------------------------------------------------------------------
__global__ __launch_bounds__(256) void deg_scatter_kernel(
    const int* __restrict__ eidx, const float* __restrict__ ew,
    float* __restrict__ deg, int E) {
    const int e = blockIdx.x * blockDim.x + threadIdx.x;
    if (e < E) atomicAdd(&deg[eidx[E + e]], ew[e]);
}

// ---------------------------------------------------------------------------
// dinv = rsqrt(deg), in place (deg >= 1 always)
// ---------------------------------------------------------------------------
__global__ __launch_bounds__(256) void rsqrt_kernel(
    float* __restrict__ deg_l, float* __restrict__ deg_g, int n) {
    const int i = blockIdx.x * blockDim.x + threadIdx.x;
    if (i < n) {
        deg_l[i] = rsqrtf(deg_l[i]);
        deg_g[i] = rsqrtf(deg_g[i]);
    }
}

// ---------------------------------------------------------------------------
// WMMA GEMM: y[i,:] = dinv[i] * (xa[i,:] @ W)   (xw scaled by dinv[src])
//            acc[i,:] += dinv[i] * y[i,:]       (self-loop term dinv^2 * xw)
// One wave per 16x16 output tile; block = 8 waves = one 16-row strip of
// all 128 output columns. K=128 -> 4x v_wmma_f32_16x16x32_f16 per tile.
// Bt is W transposed: Bt[n*128 + k] so B fragments load contiguously.
// ---------------------------------------------------------------------------
__global__ __launch_bounds__(256) void gemm_wmma_kernel(
    const _Float16* __restrict__ Ah, const _Float16* __restrict__ Bt,
    const float* __restrict__ dinv,
    float* __restrict__ y, float* __restrict__ acc, int n) {
    const int lane = threadIdx.x & 31;
    const int wave = threadIdx.x >> 5;    // 0..7 -> column tile
    const int l15  = lane & 15;
    const int hi   = lane >> 4;           // half-wave group
    const int rowTile = blockIdx.x * 16;

    int arow = rowTile + l15;             // A-fragment row (M = lane&15)
    if (arow >= n) arow = n - 1;          // clamp: rows are independent in D=A*B
    const int ncol = wave * 16 + l15;     // B/D column (N = lane&15)

    const _Float16* __restrict__ Ap = Ah + (size_t)arow * FEAT;
    const _Float16* __restrict__ Bp = Bt + (size_t)ncol * FEAT;

    v8f c = {};
#pragma unroll
    for (int ks = 0; ks < FEAT; ks += 32) {
        union { v16h v; v8h h[2]; } a, b;
        // A 16x32 f16 layout: lanes 0-15 hold K=0..7 & 16..23, lanes 16-31 K=8..15 & 24..31
        const int akb = ks + hi * 8;
        a.h[0] = *(const v8h*)(Ap + akb);
        a.h[1] = *(const v8h*)(Ap + akb + 16);
        // B 32x16 f16 layout: lanes 0-15 hold K=0..7 & 16..23, lanes 16-31 K=8..15 & 24..31
        const int bkb = ks + hi * 8;
        b.h[0] = *(const v8h*)(Bp + bkb);
        b.h[1] = *(const v8h*)(Bp + bkb + 16);
        c = __builtin_amdgcn_wmma_f32_16x16x32_f16(
                /*neg_a=*/false, a.v, /*neg_b=*/false, b.v,
                /*c_mod=*/(short)0, c, /*reuse_a=*/false, /*reuse_b=*/false);
    }

    // D layout: col = lane&15, row = vgpr + 8*(lane>=16)
#pragma unroll
    for (int v = 0; v < 8; ++v) {
        const int orow = rowTile + v + 8 * hi;
        if (orow < n) {
            const float dv = dinv[orow];
            const float yv = dv * c[v];
            const size_t idx = (size_t)orow * FEAT + ncol;
            y[idx]   = yv;
            acc[idx] += dv * yv;   // unique element per thread; stream-ordered vs scatter
        }
    }
}

// ---------------------------------------------------------------------------
// edge scatter: acc[dst,:] += w_e * dinv[dst] * y[src,:]   (wave per edge)
// lane handles columns lane + {0,32,64,96} -> fully coalesced loads/atomics
// ---------------------------------------------------------------------------
__global__ __launch_bounds__(256) void edge_scatter_kernel(
    const int* __restrict__ eidx, const float* __restrict__ ew,
    const float* __restrict__ dinv, const float* __restrict__ y,
    float* __restrict__ acc, int E) {
    const int e    = (int)(((size_t)blockIdx.x * blockDim.x + threadIdx.x) >> 5);
    const int lane = threadIdx.x & 31;
    if (e >= E) return;
    const int src = eidx[e];
    const int dst = eidx[E + e];
    const float coef = ew[e] * dinv[dst];
    const float* __restrict__ ys = y + (size_t)src * FEAT;
    float* __restrict__ ad = acc + (size_t)dst * FEAT;
#pragma unroll
    for (int j = 0; j < 4; ++j) {
        const int c = lane + j * 32;
        atomicAdd(ad + c, coef * ys[c]);
    }
}

// ---------------------------------------------------------------------------
// final: wei = sigmoid(acc); out = 2a*wei + 2l*(1-wei)
// ---------------------------------------------------------------------------
__global__ __launch_bounds__(256) void final_kernel(
    const float* __restrict__ act, const float* __restrict__ lrn,
    const float* __restrict__ acc, float* __restrict__ out, size_t total) {
    const size_t i = (size_t)blockIdx.x * blockDim.x + threadIdx.x;
    if (i < total) {
        const float wei = 1.0f / (1.0f + expf(-acc[i]));
        out[i] = 2.0f * act[i] * wei + 2.0f * lrn[i] * (1.0f - wei);
    }
}

// ---------------------------------------------------------------------------
extern "C" void kernel_launch(void* const* d_in, const int* in_sizes, int n_in,
                              void* d_out, int out_size, void* d_ws, size_t ws_size,
                              hipStream_t stream) {
    (void)n_in; (void)out_size; (void)ws_size;

    const float* act = (const float*)d_in[0];
    const float* lrn = (const float*)d_in[1];
    const int*   eil = (const int*)  d_in[2];
    const float* ewl = (const float*)d_in[3];
    const int*   eig = (const int*)  d_in[4];
    const float* ewg = (const float*)d_in[5];
    const float* Wl  = (const float*)d_in[6];
    const float* bl  = (const float*)d_in[7];
    const float* Wg  = (const float*)d_in[8];
    const float* bg  = (const float*)d_in[9];
    float* out = (float*)d_out;

    const int n  = in_sizes[0] / FEAT;
    const int El = in_sizes[3];
    const int Eg = in_sizes[5];
    const size_t total = (size_t)n * FEAT;

    // workspace carve-up (all chunk sizes 256B-aligned for these shapes)
    char* ws = (char*)d_ws;
    _Float16* xa_h = (_Float16*)ws;              ws += total * sizeof(_Float16);
    _Float16* Wl_t = (_Float16*)ws;              ws += (size_t)FEAT * FEAT * sizeof(_Float16);
    _Float16* Wg_t = (_Float16*)ws;              ws += (size_t)FEAT * FEAT * sizeof(_Float16);
    float* y     = (float*)ws;                   ws += total * sizeof(float);
    float* acc   = (float*)ws;                   ws += total * sizeof(float);
    float* deg_l = (float*)ws;                   ws += (size_t)n * sizeof(float);
    float* deg_g = (float*)ws;

    const int nd_blocks = (int)((total + 255) / 256);

    prep_kernel<<<nd_blocks, 256, 0, stream>>>(act, lrn, Wl, Wg, bl, bg,
                                               xa_h, Wl_t, Wg_t, acc, deg_l, deg_g, n);

    deg_scatter_kernel<<<(El + 255) / 256, 256, 0, stream>>>(eil, ewl, deg_l, El);
    deg_scatter_kernel<<<(Eg + 255) / 256, 256, 0, stream>>>(eig, ewg, deg_g, Eg);
    rsqrt_kernel<<<(n + 255) / 256, 256, 0, stream>>>(deg_l, deg_g, n);

    const int gemm_blocks = (n + 15) / 16;

    // local graph
    gemm_wmma_kernel<<<gemm_blocks, 256, 0, stream>>>(xa_h, Wl_t, deg_l, y, acc, n);
    edge_scatter_kernel<<<(El + 7) / 8, 256, 0, stream>>>(eil, ewl, deg_l, y, acc, El);

    // global graph (reuses y; stream-ordered after local scatter)
    gemm_wmma_kernel<<<gemm_blocks, 256, 0, stream>>>(xa_h, Wg_t, deg_g, y, acc, n);
    edge_scatter_kernel<<<(Eg + 7) / 8, 256, 0, stream>>>(eig, ewg, deg_g, y, acc, Eg);

    final_kernel<<<nd_blocks, 256, 0, stream>>>(act, lrn, acc, out, total);
}
